// QuantizableLinear_55851754717325
// MI455X (gfx1250) — matrix operand order repaired and checked by reading, not compiled, not run on hardware
//
#include <hip/hip_runtime.h>
#include <hip/hip_bf16.h>

typedef __attribute__((ext_vector_type(16))) _Float16 v16h;
typedef __attribute__((ext_vector_type(2)))  _Float16 v2h;
typedef __attribute__((ext_vector_type(8)))  float    v8f;
typedef __attribute__((ext_vector_type(4)))  float    v4f;
typedef __attribute__((ext_vector_type(4)))  int      v4i;

#define K_IN   256
#define N_OUT  256
#define MTILE  128   // rows per block (8 waves x 16 rows)

#if __has_builtin(__builtin_amdgcn_global_load_async_to_lds_b128) && \
    __has_builtin(__builtin_amdgcn_s_wait_asynccnt)
#define FQ_USE_ASYNC_LDS 1
#else
#define FQ_USE_ASYNC_LDS 0
#endif

// ---------------------------------------------------------------------------
// Kernel 1: per-tensor amax of weight and bias -> scales[0]=max|W|/127,
// scales[1]=max|b|/127. Single block.
// ---------------------------------------------------------------------------
__global__ __launch_bounds__(1024)
void fq_amax_kernel(const float* __restrict__ w, const float* __restrict__ b,
                    float* __restrict__ scales, int wn, int bn) {
    __shared__ float sw[1024];
    __shared__ float sb[1024];
    const int t = threadIdx.x;
    float mw = 0.0f;
    for (int i = t; i < wn; i += 1024) mw = fmaxf(mw, fabsf(w[i]));
    float mb = (t < bn) ? fabsf(b[t]) : 0.0f;
    sw[t] = mw;
    sb[t] = mb;
    __syncthreads();
    for (int s = 512; s > 0; s >>= 1) {
        if (t < s) {
            sw[t] = fmaxf(sw[t], sw[t + s]);
            sb[t] = fmaxf(sb[t], sb[t + s]);
        }
        __syncthreads();
    }
    if (t == 0) {
        scales[0] = sw[0] / 127.0f;
        scales[1] = sb[0] / 127.0f;
    }
}

// ---------------------------------------------------------------------------
// Kernel 2: quantize weight to INTEGER-valued f16 (exact) and pack into the
// WMMA B-fragment layout (32x16 KxN per fragment, contiguous K per lane half, same order as A):
//   fragment F = nt*8 + kc   (nt: 16-col tile of N, kc: 32-row chunk of K)
//   lane l (0..31): N = nt*16 + (l&15), K base = kc*32 + (l>>4)*16
//   VGPR i holds K = base+2i (lo16), base+2i+1 (hi16)
// One u32 word per (F, lane, i). Also dequantizes bias -> qb[] (f32).
// ---------------------------------------------------------------------------
__global__ __launch_bounds__(256)
void fq_quant_pack_kernel(const float* __restrict__ w, const float* __restrict__ b,
                          const float* __restrict__ scales,
                          unsigned int* __restrict__ wpack, float* __restrict__ qb) {
    const int tid = blockIdx.x * 256 + threadIdx.x;   // 0 .. 32767
    const float sw = scales[0];
    const int F    = tid >> 8;        // fragment index (0..127)
    const int rem  = tid & 255;
    const int lane = rem >> 3;        // 0..31
    const int i    = rem & 7;         // VGPR index 0..7
    const int nt = F >> 3;
    const int kc = F & 7;
    const int n = nt * 16 + (lane & 15);
    const int k = kc * 32 + (lane >> 4) * 16 + 2 * i;
    const float w0 = w[n * K_IN + k];
    const float w1 = w[n * K_IN + k + 1];
    const float q0 = fminf(fmaxf(rintf(w0 / sw), -127.0f), 127.0f);
    const float q1 = fminf(fmaxf(rintf(w1 / sw), -127.0f), 127.0f);
    v2h p;
    p[0] = (_Float16)q0;   // exact: integer in [-127,127]
    p[1] = (_Float16)q1;
    wpack[tid] = __builtin_bit_cast(unsigned int, p);

    if (tid < N_OUT) {
        const float sb = scales[1];
        qb[tid] = fminf(fmaxf(rintf(b[tid] / sb), -127.0f), 127.0f) * sb;
    }
}

// ---------------------------------------------------------------------------
// Kernel 3: GEMM. Each wave: 16-row strip of x, full K=256 kept in registers
// as f16 hi/lo A-fragments (16x32, contiguous K per lane half). Full packed weight
// (128 KB) is staged once into LDS (async DMA if available), then 16 ntiles
// x 8 kchunks, two chained v_wmma_f32_16x16x32_f16 each (C += lo*B; C += hi*B).
// Epilogue: out = C*scale_w + qb[n].
// ---------------------------------------------------------------------------
__global__ __launch_bounds__(256)
void fq_gemm_kernel(const float* __restrict__ x, const uint4* __restrict__ wpack,
                    const float* __restrict__ qb, const float* __restrict__ scales,
                    float* __restrict__ out) {
    __shared__ uint4 ldsu[8192];   // 128 KB: 128 fragments of 1KB (full weight)

    const int tid  = threadIdx.x;
    const int wave = tid >> 5;
    const int lane = tid & 31;
    const int r    = lane & 15;    // row within 16-row subtile / col within ntile
    const int kh   = lane >> 4;    // lane-half selects K sub-range
    const long m0  = (long)blockIdx.x * MTILE + (long)wave * 16;
    const float scale = scales[0];

    // ---- Kick off the LDS weight fill (overlaps with A loads/converts) ----
#if FQ_USE_ASYNC_LDS
    {
        typedef __attribute__((address_space(1))) v4i gv4i_t;
        typedef __attribute__((address_space(3))) v4i lv4i_t;
        uint4* wp = const_cast<uint4*>(wpack);
        #pragma unroll
        for (int j = 0; j < 32; ++j) {
            __builtin_amdgcn_global_load_async_to_lds_b128(
                (gv4i_t*)(wp + tid + j * 256),
                (lv4i_t*)(ldsu + tid + j * 256),
                0, 0);
        }
    }
#else
    #pragma unroll
    for (int j = 0; j < 32; ++j)
        ldsu[tid + j * 256] = wpack[tid + j * 256];
#endif

    // ---- Load A fragments for the whole K=256: 8 chunks, hi/lo split ----
    v16h ah[8], al[8];
    const float* xrow = x + (m0 + r) * K_IN;
    #pragma unroll
    for (int kc = 0; kc < 8; ++kc) {
        const float* p = xrow + kc * 32 + kh * 16;
        v4f f0 = *(const v4f*)(p);
        v4f f1 = *(const v4f*)(p + 4);
        v4f f2 = *(const v4f*)(p + 8);
        v4f f3 = *(const v4f*)(p + 12);
        float f[16];
        #pragma unroll
        for (int j = 0; j < 4; ++j) {
            f[j]      = f0[j];
            f[j + 4]  = f1[j];
            f[j + 8]  = f2[j];
            f[j + 12] = f3[j];
        }
        #pragma unroll
        for (int j = 0; j < 16; ++j) {
            const float v   = f[j];
            const _Float16 h = (_Float16)v;
            ah[kc][j] = h;
            al[kc][j] = (_Float16)(v - (float)h);
        }
    }

#if FQ_USE_ASYNC_LDS
    __builtin_amdgcn_s_wait_asynccnt(0);
#endif
    __syncthreads();

    // ---- Main compute: 16 ntiles x 8 kchunks, 2 WMMAs each ----
    #pragma unroll 1
    for (int nt = 0; nt < 16; ++nt) {
        v8f c = {};
        #pragma unroll
        for (int kc = 0; kc < 8; ++kc) {
            const v16h bm = *(const v16h*)(ldsu + (nt * 8 + kc) * 64 + lane * 2);
            c = __builtin_amdgcn_wmma_f32_16x16x32_f16(
                    false, al[kc], false, bm, (short)0, c, false, false);
            c = __builtin_amdgcn_wmma_f32_16x16x32_f16(
                    false, ah[kc], false, bm, (short)0, c, false, false);
        }
        const int n0 = nt * 16;
        const float qbv = qb[n0 + r];
        // C/D layout: VGPR v -> row m0 + v + 8*kh, col n0 + r
        float* orow = out + (m0 + kh * 8) * N_OUT + n0 + r;
        #pragma unroll
        for (int v = 0; v < 8; ++v)
            orow[(long)v * N_OUT] = c[v] * scale + qbv;
    }
}

// ---------------------------------------------------------------------------
// Launcher. Workspace layout (bytes):
//   [0,8)      : scales (2 f32)
//   [256,1280) : qb (256 f32)
//   [2048, 2048+131072) : packed integer-f16 weight fragments
// ---------------------------------------------------------------------------
extern "C" void kernel_launch(void* const* d_in, const int* in_sizes, int n_in,
                              void* d_out, int out_size, void* d_ws, size_t ws_size,
                              hipStream_t stream) {
    const float* x = (const float*)d_in[0];
    const float* w = (const float*)d_in[1];
    const float* b = (const float*)d_in[2];
    float* out = (float*)d_out;

    float* scales = (float*)d_ws;
    float* qb     = (float*)d_ws + 64;
    unsigned int* wpack = (unsigned int*)((char*)d_ws + 2048);

    const int wn = in_sizes[1];           // 65536
    const int bn = in_sizes[2];           // 256
    const int Brows = in_sizes[0] / K_IN; // 524288

    fq_amax_kernel<<<1, 1024, 0, stream>>>(w, b, scales, wn, bn);
    fq_quant_pack_kernel<<<128, 256, 0, stream>>>(w, b, scales, wpack, qb);
    fq_gemm_kernel<<<Brows / MTILE, 256, 0, stream>>>(
        x, (const uint4*)wpack, qb, scales, out);
}
